// SharpCosSim2D_89326729822460
// MI455X (gfx1250) — matrix operand order repaired and checked by reading, not compiled, not run on hardware
//
#include <hip/hip_runtime.h>
#include <math.h>

#define EPS   1e-7f
#define BSZ   16
#define HH    112
#define WW    112
#define CC    64
#define FF    128
#define KK    576     // 9 * 64
#define KSTEP 18      // 576 / 32
#define MT    4       // M sub-tiles (16 pixels each) per block
#define FRAG  (KSTEP * 32 * 16)   // bf16 elements per A sub-tile

typedef __attribute__((ext_vector_type(16))) __bf16 v16bf;
typedef __attribute__((ext_vector_type(8)))  float  v8f;

// scalar f32 -> bf16 bits, round-to-nearest-even
static __device__ __forceinline__ unsigned int f2bf1(float f) {
    unsigned int u = __float_as_uint(f);
    u += 0x7FFFu + ((u >> 16) & 1u);
    return u >> 16;
}

// two f32 -> packed bf16x2 dword (a in low half, b in high half)
static __device__ __forceinline__ unsigned int cvt_pk_bf16(float a, float b) {
#if __has_builtin(__builtin_amdgcn_cvt_pk_bf16_f32)
    return __builtin_bit_cast(unsigned int,
                              __builtin_amdgcn_cvt_pk_bf16_f32(a, b));
#else
    return f2bf1(a) | (f2bf1(b) << 16);
#endif
}

// positive-base fast pow: exp2(y * log2(x)), x > 0 guaranteed (|v|+EPS)
static __device__ __forceinline__ float fast_pow(float x, float y) {
#if __has_builtin(__builtin_amdgcn_logf) && __has_builtin(__builtin_amdgcn_exp2f)
    return __builtin_amdgcn_exp2f(y * __builtin_amdgcn_logf(x));
#else
    return __powf(x, y);
#endif
}

// ---------------------------------------------------------------------------
// Prep: w_norm over K, normalize, bf16 hi/lo split, swizzle into WMMA
// B-fragment layout: lane = n + (kk>=16 ? 16 : 0), elem = kk & 15.
// One block per filter f.
// ---------------------------------------------------------------------------
__global__ __launch_bounds__(256) void prep_weights(
    const float* __restrict__ w, unsigned short* __restrict__ bhiP,
    unsigned short* __restrict__ bloP)
{
    __shared__ float red[256];
    const int f   = blockIdx.x;
    const int tid = threadIdx.x;

    float s = 0.f;
    for (int k = tid; k < KK; k += 256) {
        float v = w[k * FF + f];
        s += v * v;
    }
    red[tid] = s;
    __syncthreads();
    for (int off = 128; off > 0; off >>= 1) {
        if (tid < off) red[tid] += red[tid + off];
        __syncthreads();
    }
    const float inv_wn = 1.0f / sqrtf(fmaxf(red[0], EPS));

    for (int k = tid; k < KK; k += 256) {
        float v = w[k * FF + f] * inv_wn;
        unsigned int hi = f2bf1(v);
        unsigned int lo = f2bf1(v - __uint_as_float(hi << 16));
        int snum = k >> 5, kk = k & 31;
        int lane = (f & 15) + ((kk & 16) ? 16 : 0);
        int e    = kk & 15;
        int idx  = (((((f >> 4) * KSTEP) + snum) * 32 + lane) << 4) + e;
        bhiP[idx] = (unsigned short)hi;
        bloP[idx] = (unsigned short)lo;
    }
}

// ---------------------------------------------------------------------------
// Main: block = 64 pixels (4 M-tiles of 16), 8 waves = 8 N-tiles of 16
// filters. Per K-step each wave loads its B hi/lo fragments ONCE and reuses
// them across 4 A-tiles: 12 wmma / step on 4 independent accumulator chains.
// ---------------------------------------------------------------------------
__global__ __launch_bounds__(256) void sharp_cos_main(
    const float* __restrict__ img,
    const unsigned short* __restrict__ bhiP,
    const unsigned short* __restrict__ bloP,
    const float* __restrict__ bias,
    const float* __restrict__ pexp,
    const float* __restrict__ qoff,
    float* __restrict__ out)
{
    __shared__ __align__(32) unsigned short aHi[MT * FRAG];   // 73728 B
    __shared__ __align__(32) unsigned short aLo[MT * FRAG];   // 73728 B
    __shared__ float partial[256];
    __shared__ float xinv[MT * 16];

    const int tid     = threadIdx.x;
    const int pixbase = blockIdx.x * (MT * 16);
    const float q     = qoff[0];

    // ---- build 64x576 patch in LDS, pre-swizzled to A-fragment layout ----
    // A layout: lane = m + (kk>=16 ? 16 : 0), elem = kk & 15
    // Thread owns 4 contiguous channels of one pixel per 3x3 tap:
    // one float4 global load, packed bf16 cvt, two ds_store_b64.
    const int m   = tid >> 4;        // pixel within 16-row sub-tile
    const int c4  = (tid & 15) << 2; // channel group base (0,4,...,60)

    for (int ppi = 0; ppi < MT; ++ppi) {
        const int pg   = pixbase + ppi * 16 + m;
        const int bimg = pg / (HH * WW);
        const int rem  = pg - bimg * (HH * WW);
        const int yy   = rem / WW;
        const int xx   = rem - yy * WW;

        float ssum = 0.f;
#pragma unroll
        for (int tap = 0; tap < 9; ++tap) {
            int di = tap / 3, dj = tap - di * 3;
            int sy = yy + di - 1, sx = xx + dj - 1;
            float4 v4 = make_float4(0.f, 0.f, 0.f, 0.f);
            if (sy >= 0 && sy < HH && sx >= 0 && sx < WW)
                v4 = *(const float4*)&img[(((bimg * HH + sy) * WW + sx) << 6) + c4];
            ssum += v4.x * v4.x + v4.y * v4.y + v4.z * v4.z + v4.w * v4.w;

            unsigned int h01 = cvt_pk_bf16(v4.x, v4.y);
            unsigned int h23 = cvt_pk_bf16(v4.z, v4.w);
            float r0 = v4.x - __uint_as_float(h01 << 16);
            float r1 = v4.y - __uint_as_float(h01 & 0xFFFF0000u);
            float r2 = v4.z - __uint_as_float(h23 << 16);
            float r3 = v4.w - __uint_as_float(h23 & 0xFFFF0000u);
            unsigned int l01 = cvt_pk_bf16(r0, r1);
            unsigned int l23 = cvt_pk_bf16(r2, r3);

            int k     = (tap << 6) + c4;        // 4-aligned
            int s     = k >> 5, kk = k & 31;
            int flane = m + ((kk & 16) ? 16 : 0);
            int e     = kk & 15;   // in {0,4,8,12}
            int idx   = ppi * FRAG + (((s << 5) + flane) << 4) + e;
            *(uint2*)(aHi + idx) = make_uint2(h01, h23);
            *(uint2*)(aLo + idx) = make_uint2(l01, l23);
        }
        partial[tid] = ssum;
        __syncthreads();
        if (tid < 16) {
            float s = 0.f;
#pragma unroll
            for (int i = 0; i < 16; ++i) s += partial[(tid << 4) + i];
            xinv[ppi * 16 + tid] = 1.0f / (sqrtf(fmaxf(s, EPS)) + q * q);
        }
        __syncthreads();
    }

    // ---- GEMM: K=576 in 18 steps; B loaded once/step, reused by 4 A-tiles
    const int lane = tid & 31;
    const int t    = tid >> 5;                  // N-tile (filter group)

    v8f acc0 = {0.f,0.f,0.f,0.f,0.f,0.f,0.f,0.f};
    v8f acc1 = acc0, acc2 = acc0, acc3 = acc0;

    const unsigned short* bh = bhiP + (((t * KSTEP) * 32 + lane) << 4);
    const unsigned short* bl = bloP + (((t * KSTEP) * 32 + lane) << 4);
    const unsigned short* ah = aHi + (lane << 4);
    const unsigned short* al = aLo + (lane << 4);

#pragma unroll 1
    for (int s = 0; s < KSTEP; ++s) {
        v16bf Bh = *(const v16bf*)bh;
        v16bf Bl = *(const v16bf*)bl;

        v16bf Ah0 = *(const v16bf*)(ah + 0 * FRAG);
        v16bf Al0 = *(const v16bf*)(al + 0 * FRAG);
        acc0 = __builtin_amdgcn_wmma_f32_16x16x32_bf16(false, Al0, false, Bh, (short)0, acc0, false, false);
        acc0 = __builtin_amdgcn_wmma_f32_16x16x32_bf16(false, Ah0, false, Bl, (short)0, acc0, false, false);
        acc0 = __builtin_amdgcn_wmma_f32_16x16x32_bf16(false, Ah0, false, Bh, (short)0, acc0, false, false);

        v16bf Ah1 = *(const v16bf*)(ah + 1 * FRAG);
        v16bf Al1 = *(const v16bf*)(al + 1 * FRAG);
        acc1 = __builtin_amdgcn_wmma_f32_16x16x32_bf16(false, Al1, false, Bh, (short)0, acc1, false, false);
        acc1 = __builtin_amdgcn_wmma_f32_16x16x32_bf16(false, Ah1, false, Bl, (short)0, acc1, false, false);
        acc1 = __builtin_amdgcn_wmma_f32_16x16x32_bf16(false, Ah1, false, Bh, (short)0, acc1, false, false);

        v16bf Ah2 = *(const v16bf*)(ah + 2 * FRAG);
        v16bf Al2 = *(const v16bf*)(al + 2 * FRAG);
        acc2 = __builtin_amdgcn_wmma_f32_16x16x32_bf16(false, Al2, false, Bh, (short)0, acc2, false, false);
        acc2 = __builtin_amdgcn_wmma_f32_16x16x32_bf16(false, Ah2, false, Bl, (short)0, acc2, false, false);
        acc2 = __builtin_amdgcn_wmma_f32_16x16x32_bf16(false, Ah2, false, Bh, (short)0, acc2, false, false);

        v16bf Ah3 = *(const v16bf*)(ah + 3 * FRAG);
        v16bf Al3 = *(const v16bf*)(al + 3 * FRAG);
        acc3 = __builtin_amdgcn_wmma_f32_16x16x32_bf16(false, Al3, false, Bh, (short)0, acc3, false, false);
        acc3 = __builtin_amdgcn_wmma_f32_16x16x32_bf16(false, Ah3, false, Bl, (short)0, acc3, false, false);
        acc3 = __builtin_amdgcn_wmma_f32_16x16x32_bf16(false, Ah3, false, Bh, (short)0, acc3, false, false);

        ah += 32 * 16; al += 32 * 16; bh += 32 * 16; bl += 32 * 16;
    }

    // ---- epilogue: *inv_norm, sharpen (exp2/log2), bias ------------------
    const int n  = (lane & 15) + (t << 4);
    float pp = pexp[n]; pp *= pp;
    const float bb = bias[n];
    const int mb = (lane < 16) ? 0 : 8;

    v8f accs[MT] = {acc0, acc1, acc2, acc3};
#pragma unroll
    for (int mt = 0; mt < MT; ++mt) {
#pragma unroll
        for (int r = 0; r < 8; ++r) {
            int mm  = mt * 16 + mb + r;
            float v = accs[mt][r] * xinv[mm];
            float sg = (v > 0.f) ? 1.f : ((v < 0.f) ? -1.f : 0.f);
            out[(size_t)(pixbase + mm) * FF + n] =
                sg * fast_pow(fabsf(v) + EPS, pp) + bb;
        }
    }
}

// ---------------------------------------------------------------------------
extern "C" void kernel_launch(void* const* d_in, const int* in_sizes, int n_in,
                              void* d_out, int out_size, void* d_ws, size_t ws_size,
                              hipStream_t stream)
{
    const float* img  = (const float*)d_in[0];
    const float* w    = (const float*)d_in[1];
    const float* bias = (const float*)d_in[2];
    const float* p    = (const float*)d_in[3];
    const float* q    = (const float*)d_in[4];

    unsigned short* bhiP = (unsigned short*)d_ws;       // 128*576 bf16
    unsigned short* bloP = bhiP + (size_t)FF * KK;      // 128*576 bf16

    prep_weights<<<FF, 256, 0, stream>>>(w, bhiP, bloP);

    const int mtiles = (BSZ * HH * WW) / (MT * 16);     // 3136 blocks
    sharp_cos_main<<<mtiles, 256, 0, stream>>>(img, bhiP, bloP, bias, p, q,
                                               (float*)d_out);
}
